// MyGRU_22832046146055
// MI455X (gfx1250) — matrix operand order repaired and checked by reading, not compiled, not run on hardware
//
#include <hip/hip_runtime.h>
#include <hip/hip_bf16.h>
#include <math.h>

#define S_LEN 128
#define BATCH 256
#define IDIM  512
#define HDIM  512

// 32-elem k-slice tiles: 64B data + 16B pad = 80B pitch (async-staged A tiles)
#define TPITCH 40
// full 512-elem rows staged by TDM with pad_enable: 1024B + 16B = 1040B pitch
// (260 dwords; 260 mod 64 = 4 -> conflict-free b128 fragment reads)
#define WPITCH 520

typedef __attribute__((ext_vector_type(16))) __bf16    v16bf;
typedef __attribute__((ext_vector_type(8)))  float     v8f;
typedef __attribute__((ext_vector_type(4)))  unsigned  v4u;
typedef __attribute__((ext_vector_type(8)))  unsigned  v8u;

union FragB16 { uint4 u[2]; v16bf v; };

__device__ __forceinline__ unsigned short f32_to_bf16_bits(float f) {
  union { float f; unsigned int u; } c; c.f = f;
  unsigned int r = c.u + 0x7FFFu + ((c.u >> 16) & 1u);
  return (unsigned short)(r >> 16);
}

// ---- Tensor Data Mover: 2-D tile DMA global -> LDS (TENSOR_LOAD_TO_LDS) ----
// D# per CDNA5 ISA. 2-group form (<=2D tensor).
// Elements are 2-byte (bf16). Padding: pad_interval=7 (256 dwords = one
// 512-elem row), pad_amount=3 (4 dwords) -> 1040B LDS row pitch.
__device__ __forceinline__ void tdm_load_2d_bf16(unsigned ldsAddr, unsigned long long gaddr,
                                                 unsigned tensorD0, unsigned tensorD1,
                                                 unsigned tileD0, unsigned tileD1,
                                                 unsigned stride0) {
  v4u g0;
  g0.x = 1u;                                          // count=1 (valid), user mode
  g0.y = ldsAddr;                                     // lds_addr[31:0]
  g0.z = (unsigned)(gaddr & 0xFFFFFFFFull);           // global_addr[31:0]
  g0.w = (unsigned)((gaddr >> 32) & 0x01FFFFFFull)    // global_addr[56:32]
       | (2u << 30);                                  // type=2 ("image")
  v8u g1;
  g1.s0 = (1u << 16)                                  // data_size=1 -> 2 bytes
        | (1u << 20)                                  // pad_enable
        | (7u << 22)                                  // pad_interval: 256 dwords
        | (3u << 25);                                 // pad_amount: 4 dwords (16B)
  g1.s1 = (tensorD0 & 0xFFFFu) << 16;                 // tensor_dim0[15:0] @ bit48
  g1.s2 = (tensorD0 >> 16) | ((tensorD1 & 0xFFFFu) << 16);
  g1.s3 = (tensorD1 >> 16) | (tileD0 << 16);          // tile_dim0 @ bit112
  g1.s4 = tileD1 & 0xFFFFu;                           // tile_dim1; tile_dim2=0
  g1.s5 = stride0;                                    // tensor_dim0_stride[31:0]
  g1.s6 = 0u;
  g1.s7 = 0u;
  asm volatile("tensor_load_to_lds %0, %1" :: "s"(g0), "s"(g1) : "memory");
}

// ---- async memory -> LDS staging (GLOBAL_LOAD_ASYNC_TO_LDS_B128) ----------
__device__ __forceinline__ void async_copy_tile(const unsigned short* __restrict__ g,
                                                int ldg, int rows, int kBase,
                                                unsigned short* sm, int tid, int nthr) {
  int chunks = rows * 4;  // 4 x 16B per 32-elem row
  for (int c = tid; c < chunks; c += nthr) {
    int row = c >> 2, cw = c & 3;
    const unsigned short* gp = g + (size_t)row * ldg + kBase + cw * 8;
    unsigned lds = (unsigned)(unsigned long long)(uintptr_t)(sm + (size_t)row * TPITCH + cw * 8);
    unsigned long long ga = (unsigned long long)(uintptr_t)gp;
    asm volatile("global_load_async_to_lds_b128 %0, %1, off"
                 :: "v"(lds), "v"(ga) : "memory");
  }
}
__device__ __forceinline__ void wait_async_all() {
  asm volatile("s_wait_asynccnt 0" ::: "memory");
}

// ---- WMMA fragment loads --------------------------------------------------
// A-frag 16x32 from 80B-pitch tile
__device__ __forceinline__ v16bf lds_a_frag(const unsigned short* sm, int mBase, int lane) {
  int hf = lane >> 4, m = lane & 15;
  const unsigned short* p = sm + (size_t)(mBase + m) * TPITCH + hf * 8;
  FragB16 f;
  f.u[0] = *(const uint4*)(p);
  f.u[1] = *(const uint4*)(p + 16);
  return f.v;
}
// A-frag 16x32 from TDM-staged 1040B-pitch slab (k offset kBase)
__device__ __forceinline__ v16bf lds_a_frag_p(const unsigned short* sm, int mBase, int kBase, int lane) {
  int hf = lane >> 4, m = lane & 15;
  const unsigned short* p = sm + (size_t)(mBase + m) * WPITCH + kBase + hf * 8;
  FragB16 f;
  f.u[0] = *(const uint4*)(p);
  f.u[1] = *(const uint4*)(p + 16);
  return f.v;
}
// B-frag 32x16 (W^T) from TDM-staged 1040B-pitch slab
__device__ __forceinline__ v16bf lds_b_frag_p(const unsigned short* sm, int nBase, int kBase, int lane) {
  int hf = lane >> 4, n = lane & 15;
  const unsigned short* p = sm + (size_t)(nBase + n) * WPITCH + kBase + hf * 8;
  FragB16 f;
  f.u[0] = *(const uint4*)(p);
  f.u[1] = *(const uint4*)(p + 16);
  return f.v;
}
// Direct-global B-fragment (L2-resident recurrent weights)
__device__ __forceinline__ v16bf glb_b_frag(const unsigned short* __restrict__ W,
                                            int ldb, int nBase, int kBase, int lane) {
  int hf = lane >> 4, n = lane & 15;
  const unsigned short* p = W + (size_t)(nBase + n) * ldb + kBase + hf * 8;
  FragB16 f;
  f.u[0] = *(const uint4*)(p);
  f.u[1] = *(const uint4*)(p + 16);
  return f.v;
}

// ---------------------------------------------------------------------------
__global__ void cvt_bf16(const float* __restrict__ src, unsigned short* __restrict__ dst, int n) {
  int i = blockIdx.x * blockDim.x + threadIdx.x;
  if (i < n) dst[i] = f32_to_bf16_bits(src[i]);
}

__global__ void init_state(const float* __restrict__ h0, unsigned short* __restrict__ hB,
                           float* __restrict__ hF, unsigned int* __restrict__ ctr, int n) {
  int i = blockIdx.x * blockDim.x + threadIdx.x;
  if (i == 0) *ctr = 0u;
  if (i < n) { float v = h0[i]; hB[i] = f32_to_bf16_bits(v); hF[i] = v; }
}

// ---------------------------------------------------------------------------
// Input projections X_g = input @ W_g^T + bias (b_hr/b_hz folded in).
// M=32768, N=512, K=512. Block 256 thr = 8 waves (4x2), tile 256x128.
// B weight slab (128x512) staged ONCE per block by the TDM (padded pitch);
// A k-tiles double-buffered via async-to-LDS.
__global__ __launch_bounds__(256) void xproj_gemm(
    const unsigned short* __restrict__ Abf,
    const unsigned short* __restrict__ Wir,
    const unsigned short* __restrict__ Wiz,
    const unsigned short* __restrict__ Win,
    const float* __restrict__ b_ir, const float* __restrict__ b_hr,
    const float* __restrict__ b_iz, const float* __restrict__ b_hz,
    const float* __restrict__ b_in,
    float* __restrict__ Xr, float* __restrict__ Xz, float* __restrict__ Xn) {
  __shared__ unsigned short smA[2][256 * TPITCH];  // 2 x 20KB
  __shared__ unsigned short smW[128 * WPITCH];     // 130KB TDM-staged weights
  const int K = IDIM, N = HDIM;
  const int tid = threadIdx.x;
  const int lane = tid & 31, w = tid >> 5;
  const int nl = lane & 15, hf = lane >> 4;
  const int mL = (w >> 1) * 64;
  const int nL = (w & 1) * 64;
  const int mB = blockIdx.x * 256;
  const int nB = blockIdx.y * 128;
  const int gate = blockIdx.z;
  const unsigned short* W = (gate == 0) ? Wir : (gate == 1) ? Wiz : Win;
  float* X = (gate == 0) ? Xr : (gate == 1) ? Xz : Xn;
  const unsigned short* Ag = Abf + (size_t)mB * K;
  const unsigned short* Wg = W + (size_t)nB * K;

  v8f acc[4][4];
#pragma unroll
  for (int ni = 0; ni < 4; ++ni) {
    int col = nB + nL + ni * 16 + nl;
    float bv = (gate == 0) ? (b_ir[col] + b_hr[col])
             : (gate == 1) ? (b_iz[col] + b_hz[col])
             : b_in[col];
#pragma unroll
    for (int mi = 0; mi < 4; ++mi)
#pragma unroll
      for (int r = 0; r < 8; ++r) acc[mi][ni][r] = bv;
  }

  // wave 0 DMAs the whole 128x512 weight slab (TDM, padded rows)
  if (w == 0) {
    tdm_load_2d_bf16((unsigned)(uintptr_t)&smW[0], (unsigned long long)(uintptr_t)Wg,
                     K, 128, K, 128, K);
    __builtin_amdgcn_s_wait_tensorcnt(0);
  }
  // prologue: stage A k-tile 0
  async_copy_tile(Ag, K, 256, 0, smA[0], tid, 256);

  const int NKT = K / 32;
  for (int kt = 0; kt < NKT; ++kt) {
    wait_async_all();
    __syncthreads();  // A tile resident + (iter 0) TDM slab visible
    if (kt + 1 < NKT)
      async_copy_tile(Ag, K, 256, (kt + 1) * 32, smA[(kt + 1) & 1], tid, 256);
    const unsigned short* sA = smA[kt & 1];
    const int k0 = kt * 32;
    v16bf a[4], b[4];
#pragma unroll
    for (int mi = 0; mi < 4; ++mi) a[mi] = lds_a_frag(sA, mL + mi * 16, lane);
#pragma unroll
    for (int ni = 0; ni < 4; ++ni) b[ni] = lds_b_frag_p(smW, nL + ni * 16, k0, lane);
#pragma unroll
    for (int mi = 0; mi < 4; ++mi)
#pragma unroll
      for (int ni = 0; ni < 4; ++ni)
        acc[mi][ni] = __builtin_amdgcn_wmma_f32_16x16x32_bf16(
            false, a[mi], false, b[ni], (short)0, acc[mi][ni], false, false);
  }

#pragma unroll
  for (int mi = 0; mi < 4; ++mi)
#pragma unroll
    for (int ni = 0; ni < 4; ++ni) {
      int col = nB + nL + ni * 16 + nl;
      int row0 = mB + mL + mi * 16 + hf * 8;
#pragma unroll
      for (int r = 0; r < 8; ++r)
        X[(size_t)(row0 + r) * N + col] = acc[mi][ni][r];
    }
}

// ---------------------------------------------------------------------------
// Persistent recurrent scan: 8 workgroups (2 M x 4 N tiles of 128x128),
// 8 waves (2x4), wave tile 64x32, 3 gates fused. Each step, the TDM DMAs the
// block's entire 128x512 h slab into padded LDS (one tensor_load_to_lds, one
// s_wait_tensorcnt, one barrier) -> barrier-free k-loop. Weights stay in L2.
__global__ __launch_bounds__(256) void gru_scan(
    const unsigned short* __restrict__ Whr,
    const unsigned short* __restrict__ Whz,
    const unsigned short* __restrict__ Whn,
    const float* __restrict__ b_hn,
    const float* __restrict__ Xr, const float* __restrict__ Xz, const float* __restrict__ Xn,
    const int* __restrict__ mask,
    unsigned short* __restrict__ hB0, unsigned short* __restrict__ hB1,
    float* __restrict__ hF0, float* __restrict__ hF1,
    float* __restrict__ out,
    unsigned int* __restrict__ ctr) {
  __shared__ unsigned short smH[128 * WPITCH];  // 130KB TDM-staged h slab
  const int H = HDIM, B = BATCH;
  const int tid = threadIdx.x;
  const int lane = tid & 31, w = tid >> 5;
  const int nl = lane & 15, hf = lane >> 4;
  const int blockM = blockIdx.x & 1;
  const int blockN = blockIdx.x >> 1;
  const int mLocal = (w >> 2) * 64;
  const int nWave = blockN * 128 + (w & 3) * 32;

#pragma unroll 1
  for (int s = 0; s < S_LEN; ++s) {
    const unsigned short* hRd = (s & 1) ? hB1 : hB0;
    unsigned short* hWr = (s & 1) ? hB0 : hB1;
    const float* hOld = (s & 1) ? hF1 : hF0;
    float* hNew = (s & 1) ? hF0 : hF1;
    const unsigned short* Ag = hRd + (size_t)(blockM * 128) * H;

    v8f accR[4][2], accZ[4][2], accN[4][2];
#pragma unroll
    for (int ni = 0; ni < 2; ++ni) {
      float bn = b_hn[nWave + ni * 16 + nl];
#pragma unroll
      for (int mi = 0; mi < 4; ++mi)
#pragma unroll
        for (int r = 0; r < 8; ++r) {
          accR[mi][ni][r] = 0.0f;
          accZ[mi][ni][r] = 0.0f;
          accN[mi][ni][r] = bn;  // b_hn sits inside the r*(...) term
        }
    }

    // one TDM DMA of the whole h slab for this step
    if (w == 0) {
      tdm_load_2d_bf16((unsigned)(uintptr_t)&smH[0], (unsigned long long)(uintptr_t)Ag,
                       H, 128, H, 128, H);
      __builtin_amdgcn_s_wait_tensorcnt(0);
    }
    __syncthreads();

    const int NKT = H / 32;
#pragma unroll 1
    for (int kt = 0; kt < NKT; ++kt) {
      const int k0 = kt * 32;
      v16bf a[4];
#pragma unroll
      for (int mi = 0; mi < 4; ++mi) a[mi] = lds_a_frag_p(smH, mLocal + mi * 16, k0, lane);
#pragma unroll
      for (int ni = 0; ni < 2; ++ni) {
        v16bf br = glb_b_frag(Whr, H, nWave + ni * 16, k0, lane);
        v16bf bz = glb_b_frag(Whz, H, nWave + ni * 16, k0, lane);
        v16bf bn = glb_b_frag(Whn, H, nWave + ni * 16, k0, lane);
#pragma unroll
        for (int mi = 0; mi < 4; ++mi) {
          accR[mi][ni] = __builtin_amdgcn_wmma_f32_16x16x32_bf16(
              false, a[mi], false, br, (short)0, accR[mi][ni], false, false);
          accZ[mi][ni] = __builtin_amdgcn_wmma_f32_16x16x32_bf16(
              false, a[mi], false, bz, (short)0, accZ[mi][ni], false, false);
          accN[mi][ni] = __builtin_amdgcn_wmma_f32_16x16x32_bf16(
              false, a[mi], false, bn, (short)0, accN[mi][ni], false, false);
        }
      }
    }

    // fused GRU cell elementwise on C-layout fragments
    const size_t xoff = (size_t)s * B * H;
#pragma unroll
    for (int mi = 0; mi < 4; ++mi)
#pragma unroll
      for (int ni = 0; ni < 2; ++ni) {
        int col = nWave + ni * 16 + nl;
        int row0 = blockM * 128 + mLocal + mi * 16 + hf * 8;
#pragma unroll
        for (int r = 0; r < 8; ++r) {
          int row = row0 + r;
          size_t ix = (size_t)row * H + col;
          float xr = Xr[xoff + ix];
          float xz = Xz[xoff + ix];
          float xn = Xn[xoff + ix];
          float ho = hOld[ix];
          float rr = fmaxf(xr + accR[mi][ni][r], 0.0f);
          float zz = fmaxf(xz + accZ[mi][ni][r], 0.0f);
          float nn = tanhf(xn + rr * accN[mi][ni][r]);
          float mv = (float)mask[s * B + row];
          float hv = ((1.0f - zz) * ho + zz * nn) * mv;
          out[(size_t)s * (B * 2 * H) + (size_t)row * (2 * H) + col] = hv;
          hNew[ix] = hv;
          hWr[ix] = f32_to_bf16_bits(hv);
        }
      }

    // device-scope step barrier (monotonic counter; reset by init_state)
    __threadfence();
    __syncthreads();
    if (threadIdx.x == 0) {
      __hip_atomic_fetch_add(ctr, 1u, __ATOMIC_RELEASE, __HIP_MEMORY_SCOPE_AGENT);
      unsigned tgt = (unsigned)(s + 1) * gridDim.x;
      while (__hip_atomic_load(ctr, __ATOMIC_ACQUIRE, __HIP_MEMORY_SCOPE_AGENT) < tgt) {
        __builtin_amdgcn_s_sleep(2);
      }
    }
    __syncthreads();
    __threadfence();
  }
}

// Second half of output = forward state at s = S-1, broadcast to every step.
__global__ void broadcast_last(float* __restrict__ out) {
  size_t idx = (size_t)blockIdx.x * blockDim.x + threadIdx.x;
  size_t total = (size_t)S_LEN * BATCH * HDIM;
  if (idx >= total) return;
  int j = (int)(idx % HDIM);
  size_t sb = idx / HDIM;
  int b = (int)(sb % BATCH);
  int s = (int)(sb / BATCH);
  float v = out[((size_t)(S_LEN - 1) * BATCH + b) * (2 * HDIM) + j];
  out[((size_t)s * BATCH + b) * (2 * HDIM) + HDIM + j] = v;
}

// ---------------------------------------------------------------------------
extern "C" void kernel_launch(void* const* d_in, const int* in_sizes, int n_in,
                              void* d_out, int out_size, void* d_ws, size_t ws_size,
                              hipStream_t stream) {
  (void)in_sizes; (void)n_in; (void)out_size; (void)ws_size;
  const float* input  = (const float*)d_in[0];
  const float* hidden = (const float*)d_in[1];
  const int*   mask   = (const int*)d_in[2];
  const float* W_ir = (const float*)d_in[3];  const float* b_ir = (const float*)d_in[4];
  const float* W_hr = (const float*)d_in[5];  const float* b_hr = (const float*)d_in[6];
  const float* W_iz = (const float*)d_in[7];  const float* b_iz = (const float*)d_in[8];
  const float* W_hz = (const float*)d_in[9];  const float* b_hz = (const float*)d_in[10];
  const float* W_in = (const float*)d_in[11]; const float* b_in = (const float*)d_in[12];
  const float* W_hn = (const float*)d_in[13]; const float* b_hn = (const float*)d_in[14];
  float* out = (float*)d_out;

  char* ws = (char*)d_ws;
  size_t off = 0;
  auto alloc = [&](size_t bytes) -> char* {
    char* p = ws + off;
    off += (bytes + 255) & ~(size_t)255;
    return p;
  };
  unsigned short* inBf = (unsigned short*)alloc((size_t)S_LEN * BATCH * IDIM * 2);
  unsigned short* WirB = (unsigned short*)alloc((size_t)HDIM * IDIM * 2);
  unsigned short* WizB = (unsigned short*)alloc((size_t)HDIM * IDIM * 2);
  unsigned short* WinB = (unsigned short*)alloc((size_t)HDIM * IDIM * 2);
  unsigned short* WhrB = (unsigned short*)alloc((size_t)HDIM * HDIM * 2);
  unsigned short* WhzB = (unsigned short*)alloc((size_t)HDIM * HDIM * 2);
  unsigned short* WhnB = (unsigned short*)alloc((size_t)HDIM * HDIM * 2);
  float* Xr = (float*)alloc((size_t)S_LEN * BATCH * HDIM * 4);
  float* Xz = (float*)alloc((size_t)S_LEN * BATCH * HDIM * 4);
  float* Xn = (float*)alloc((size_t)S_LEN * BATCH * HDIM * 4);
  unsigned short* hB0 = (unsigned short*)alloc((size_t)BATCH * HDIM * 2);
  unsigned short* hB1 = (unsigned short*)alloc((size_t)BATCH * HDIM * 2);
  float* hF0 = (float*)alloc((size_t)BATCH * HDIM * 4);
  float* hF1 = (float*)alloc((size_t)BATCH * HDIM * 4);
  unsigned int* ctr = (unsigned int*)alloc(256);

  int nin = S_LEN * BATCH * IDIM;
  cvt_bf16<<<(nin + 255) / 256, 256, 0, stream>>>(input, inBf, nin);
  int nw = HDIM * IDIM;
  cvt_bf16<<<(nw + 255) / 256, 256, 0, stream>>>(W_ir, WirB, nw);
  cvt_bf16<<<(nw + 255) / 256, 256, 0, stream>>>(W_iz, WizB, nw);
  cvt_bf16<<<(nw + 255) / 256, 256, 0, stream>>>(W_in, WinB, nw);
  cvt_bf16<<<(nw + 255) / 256, 256, 0, stream>>>(W_hr, WhrB, nw);
  cvt_bf16<<<(nw + 255) / 256, 256, 0, stream>>>(W_hz, WhzB, nw);
  cvt_bf16<<<(nw + 255) / 256, 256, 0, stream>>>(W_hn, WhnB, nw);
  int nh = BATCH * HDIM;
  init_state<<<(nh + 255) / 256, 256, 0, stream>>>(hidden, hB0, hF0, ctr, nh);

  dim3 g1(S_LEN * BATCH / 256, HDIM / 128, 3);
  xproj_gemm<<<g1, 256, 0, stream>>>(inBf, WirB, WizB, WinB,
                                     b_ir, b_hr, b_iz, b_hz, b_in,
                                     Xr, Xz, Xn);

  gru_scan<<<8, 256, 0, stream>>>(WhrB, WhzB, WhnB, b_hn, Xr, Xz, Xn, mask,
                                  hB0, hB1, hF0, hF1, out, ctr);

  size_t nbc = (size_t)S_LEN * BATCH * HDIM;
  broadcast_last<<<(unsigned)((nbc + 255) / 256), 256, 0, stream>>>(out);
}
